// RStarcoderAttention_47665547051580
// MI455X (gfx1250) — matrix operand order repaired and checked by reading, not compiled, not run on hardware
//
#include <hip/hip_runtime.h>
#include <cstdint>
#include <cstddef>

#define B_  2
#define S_  2048
#define E_  2048
#define H_  16
#define D_  128
#define NE_ 8
#define R_  16
#define HD_ (H_*D_)   // 2048
#define NR_ (NE_*R_)  // 128
#define M_  (B_*S_)   // 4096

typedef __attribute__((ext_vector_type(16))) __bf16 v16bf;
typedef __attribute__((ext_vector_type(8)))  __bf16 v8bf;
typedef __attribute__((ext_vector_type(8)))  float  v8f;
typedef __attribute__((ext_vector_type(4)))  int    v4i_;

union AFrag { v16bf v; v8bf h[2]; };

#if defined(__HIP_DEVICE_COMPILE__)
#if __has_builtin(__builtin_amdgcn_global_load_async_to_lds_b128)
#define ASYNC_LDS 1
#endif
#endif

__device__ __forceinline__ __bf16 f2bf(float f) {
  uint32_t u = __builtin_bit_cast(uint32_t, f);
  u += 0x7FFFu + ((u >> 16) & 1u);           // round-to-nearest-even
  uint16_t hs = (uint16_t)(u >> 16);
  return __builtin_bit_cast(__bf16, hs);
}

__device__ __forceinline__ v8f wmma_bf16(v16bf a, v16bf b, v8f c) {
  return __builtin_amdgcn_wmma_f32_16x16x32_bf16(false, a, false, b, (short)0, c,
                                                 false, false);
}

// A-fragment (16x32 bf16): half-wave K split {0..15}/{16..31}
__device__ __forceinline__ void loadA(AFrag& a, const __bf16* arow, int k, int half) {
  a.h[0] = *(const v8bf*)(arow + k + half * 16);
  a.h[1] = *(const v8bf*)(arow + k + half * 16 + 8);
}
// B-fragments for 4 N-subtiles: lane = column, half-wave K split {0..15}/{16..31}
__device__ __forceinline__ void loadB(v16bf* bfr, const __bf16* wbase, int k, int K,
                                      int half) {
#pragma unroll
  for (int t = 0; t < 4; ++t)
    bfr[t] = *(const v16bf*)(wbase + (size_t)(t * 16) * K + k + half * 16);
}

// ---------------------------------------------------------------------------
// elementwise f32 -> bf16 conversion (exact-size grids, all counts % 256 == 0)
// ---------------------------------------------------------------------------
__global__ __launch_bounds__(256) void cvt_f32_bf16(const float* __restrict__ in,
                                                    unsigned short* __restrict__ out) {
  size_t i = (size_t)blockIdx.x * 256 + threadIdx.x;
  out[i] = __builtin_bit_cast(unsigned short, f2bf(in[i]));
}

// ---------------------------------------------------------------------------
// C[M,N] (f32) = A[M,K](bf16) * W[N,K]^T(bf16) (+bias) (+=C)
// one wave -> 16x64 tile; K pipelined with explicit ping-pong register sets
// (2x unroll, K/32 always even) -> no rotate copies, loads overlap WMMAs.
// ---------------------------------------------------------------------------
__global__ __launch_bounds__(128) void gemm_bf16_wmma(
    const unsigned short* __restrict__ Ain, const unsigned short* __restrict__ Win,
    const float* __restrict__ bias, float* __restrict__ C,
    int Mr, int N, int K, int accumulate) {
  const __bf16* A = reinterpret_cast<const __bf16*>(Ain);
  const __bf16* W = reinterpret_cast<const __bf16*>(Win);
  int lane = threadIdx.x & 31, wave = threadIdx.x >> 5;
  int half = lane >> 4, l16 = lane & 15;
  int tiles_n4 = N >> 6;                       // 16x64 tiles along N
  int tid = blockIdx.x * 4 + wave;
  if (tid >= (Mr >> 4) * tiles_n4) return;
  int tm = tid / tiles_n4, tn4 = tid - tm * tiles_n4;

  const __bf16* arow  = A + (size_t)(tm * 16 + l16) * K;
  const __bf16* wbase = W + (size_t)(tn4 * 64 + l16) * K;

  v8f c[4] = {};
  AFrag a0, a1;
  v16bf b0[4], b1[4];
  loadA(a0, arow, 0, half);
  loadB(b0, wbase, 0, K, half);

  for (int k = 0; k < K; k += 64) {
    int k1 = k + 32;                           // always < K (K/32 even)
    int k2 = (k + 64 < K) ? k + 64 : 0;        // clamped branchless preload
    loadA(a1, arow, k1, half);
    loadB(b1, wbase, k1, K, half);
    __builtin_prefetch(arow + k + 256, 0, 1);  // global_prefetch_b8
#pragma unroll
    for (int t = 0; t < 4; ++t) c[t] = wmma_bf16(a0.v, b0[t], c[t]);
    loadA(a0, arow, k2, half);
    loadB(b0, wbase, k2, K, half);
#pragma unroll
    for (int t = 0; t < 4; ++t) c[t] = wmma_bf16(a1.v, b1[t], c[t]);
  }

  int row0 = tm * 16 + half * 8;               // C layout: VGPR g -> row g / g+8
#pragma unroll
  for (int t = 0; t < 4; ++t) {
    int col = tn4 * 64 + t * 16 + l16;
    float bi = bias ? bias[col] : 0.f;
#pragma unroll
    for (int g = 0; g < 8; ++g) {
      size_t idx = (size_t)(row0 + g) * N + col;
      float v = c[t][g] + bi;
      if (accumulate) v += C[idx];
      C[idx] = v;
    }
  }
}

// ---------------------------------------------------------------------------
// h[M,128] * mask[M,8] -> bf16 (expert gating of LoRA hidden)
// ---------------------------------------------------------------------------
__global__ __launch_bounds__(256) void mask_scale_cvt(const float* __restrict__ h,
                                                      const float* __restrict__ mask,
                                                      unsigned short* __restrict__ out) {
  int i = blockIdx.x * 256 + threadIdx.x;      // over M_*NR_
  int row = i >> 7, col = i & 127;
  float v = h[i] * mask[row * NE_ + (col >> 4)];
  out[i] = __builtin_bit_cast(unsigned short, f2bf(v));
}

// ---------------------------------------------------------------------------
// RoPE (llama-style) fused with bf16 conversion; layout [B,S,H,D]
// ---------------------------------------------------------------------------
__global__ __launch_bounds__(256) void rope_cvt(const float* __restrict__ t,
                                                unsigned short* __restrict__ out) {
  int i = blockIdx.x * 256 + threadIdx.x;
  int d = i & 127;
  int s = (i >> 11) & (S_ - 1);
  int dm = d & 63;
  float inv = __powf(10000.f, -(float)dm * (1.f / 64.f));
  float ang = (float)s * inv;
  float cs = __cosf(ang), sn = __sinf(ang);
  float rot = (d < 64) ? -t[i + 64] : t[i - 64];
  out[i] = __builtin_bit_cast(unsigned short, f2bf(t[i] * cs + rot * sn));
}

// ---------------------------------------------------------------------------
// V [B,S,H,D] f32 -> Vt [B,H,D,S] bf16  (PV B-fragments become contiguous)
// ---------------------------------------------------------------------------
__global__ __launch_bounds__(256) void v_transpose_cvt(const float* __restrict__ v,
                                                       unsigned short* __restrict__ vt) {
  int i = blockIdx.x * 256 + threadIdx.x;
  int d = i & 127;
  int hh = (i >> 7) & 15;
  int s = (i >> 11) & (S_ - 1);
  int b = i >> 22;
  size_t o = ((size_t)(b * H_ + hh) * D_ + d) * S_ + s;
  vt[o] = __builtin_bit_cast(unsigned short, f2bf(v[i]));
}

#ifdef ASYNC_LDS
// Stage a 128(d) x 32(s) bf16 V tile (8KB) into LDS with the CDNA5 async
// engine: 512 b128 chunks, 16 per lane, tracked on ASYNCcnt.
__device__ __forceinline__ void async_copy_vtile(const __bf16* vbase, int k0,
                                                 __bf16* dst, int lane) {
#pragma unroll
  for (int cch = 0; cch < 16; ++cch) {
    int chunk = cch * 32 + lane;
    int d = chunk >> 2, q = chunk & 3;
    const __bf16* g = vbase + (size_t)d * S_ + k0 + q * 8;
    __bf16* l = dst + d * 32 + q * 8;
    __builtin_amdgcn_global_load_async_to_lds_b128(
        (__attribute__((address_space(1))) v4i_*)(void*)g,
        (__attribute__((address_space(3))) v4i_*)(void*)l, 0, 0);
  }
}
#endif

// ---------------------------------------------------------------------------
// Causal flash attention: one wave per 16-row q tile, k tiles of 32,
// QK^T and PV via v_wmma_f32_16x16x32_bf16, online softmax in f32.
// V tiles double-buffered into LDS via async loads when available.
// ---------------------------------------------------------------------------
__global__ __launch_bounds__(32) void flash_attn(
    const unsigned short* __restrict__ qin, const unsigned short* __restrict__ kin,
    const unsigned short* __restrict__ vtin, unsigned short* __restrict__ oout) {
  const __bf16* qb = reinterpret_cast<const __bf16*>(qin);
  const __bf16* kb = reinterpret_cast<const __bf16*>(kin);
  const __bf16* vt = reinterpret_cast<const __bf16*>(vtin);

  int qt = blockIdx.x & 127;
  int h  = (blockIdx.x >> 7) & 15;
  int b  = blockIdx.x >> 11;
  int lane = threadIdx.x & 31;
  int half = lane >> 4, l16 = lane & 15;
  int q0 = qt * 16;

  __shared__ __attribute__((aligned(32))) __bf16 lds_p[16 * 32];
#ifdef ASYNC_LDS
  __shared__ __attribute__((aligned(32))) __bf16 lds_v[2][128 * 32];
#endif

  v16bf qf[4];                                 // Q 16x128 as 4 K-chunks of 32
  const __bf16* qrow = qb + ((size_t)((b * S_ + q0 + l16) * H_ + h) << 7);
#pragma unroll
  for (int kc = 0; kc < 4; ++kc) {
    AFrag t;
    t.h[0] = *(const v8bf*)(qrow + kc * 32 + half * 16);
    t.h[1] = *(const v8bf*)(qrow + kc * 32 + half * 16 + 8);
    qf[kc] = t.v;
  }

  v8f vzero = {};
  v8f acc[8];                                  // O 16x128 f32
#pragma unroll
  for (int j = 0; j < 8; ++j) acc[j] = vzero;
  float mr[8], lr[8];
#pragma unroll
  for (int g = 0; g < 8; ++g) { mr[g] = -3.0e38f; lr[g] = 0.f; }

  const float smul = 0.08838834764831845f;     // 1/sqrt(128)
  const __bf16* vbase = vt + ((size_t)(b * H_ + h) * D_) * S_;

#ifdef ASYNC_LDS
  async_copy_vtile(vbase, 0, &lds_v[0][0], lane);
#endif

  for (int k0 = 0; k0 <= q0 + 15; k0 += 32) {
    int ti = k0 >> 5;
    int have_next = (k0 + 32 <= q0 + 15);
#ifdef ASYNC_LDS
    if (have_next) async_copy_vtile(vbase, k0 + 32, &lds_v[(ti + 1) & 1][0], lane);
#endif
    if (have_next)                             // prefetch next K tile rows
      __builtin_prefetch(kb + ((size_t)((b * S_ + k0 + 32 + l16) * H_ + h) << 7), 0, 1);

    float p[2][8];
    float tmax[8];
#pragma unroll
    for (int g = 0; g < 8; ++g) tmax[g] = -3.0e38f;

#pragma unroll
    for (int sub = 0; sub < 2; ++sub) {        // two 16-col score subtiles
      const __bf16* krow =
          kb + ((size_t)((b * S_ + k0 + sub * 16 + l16) * H_ + h) << 7);
      v8f sc = vzero;
#pragma unroll
      for (int kc = 0; kc < 4; ++kc) {         // contract over D=128
        v16bf kfrag = *(const v16bf*)(krow + kc * 32 + half * 16);
        sc = wmma_bf16(qf[kc], kfrag, sc);
      }
      int kcol = k0 + sub * 16 + l16;
#pragma unroll
      for (int g = 0; g < 8; ++g) {
        int qr = q0 + half * 8 + g;
        float s = (kcol <= qr) ? sc[g] * smul : -__builtin_inff();
        p[sub][g] = s;
        tmax[g] = fmaxf(tmax[g], s);
      }
    }

#pragma unroll
    for (int g = 0; g < 8; ++g) {              // row max across 16 lanes (half-wave)
      float a = tmax[g];
#pragma unroll
      for (int off = 1; off < 16; off <<= 1) a = fmaxf(a, __shfl_xor(a, off, 16));
      float nm = fmaxf(mr[g], a);
      float corr = __expf(mr[g] - nm);
      mr[g] = nm;
      lr[g] *= corr;
#pragma unroll
      for (int j = 0; j < 8; ++j) acc[j][g] *= corr;
    }

#pragma unroll
    for (int sub = 0; sub < 2; ++sub)
#pragma unroll
      for (int g = 0; g < 8; ++g) {
        float e = __expf(p[sub][g] - mr[g]);
        p[sub][g] = e;
        lds_p[(half * 8 + g) * 32 + sub * 16 + l16] = f2bf(e);
      }

#pragma unroll
    for (int g = 0; g < 8; ++g) {              // row sums
      float ps = p[0][g] + p[1][g];
#pragma unroll
      for (int off = 1; off < 16; off <<= 1) ps += __shfl_xor(ps, off, 16);
      lr[g] += ps;
    }

    __syncthreads();
    AFrag pt;                                  // reload P in A 16x32 layout
    {
      const __bf16* pr = lds_p + l16 * 32;
      pt.h[0] = *(const v8bf*)(pr + half * 16);
      pt.h[1] = *(const v8bf*)(pr + half * 16 + 8);
    }

#ifdef ASYNC_LDS
    // current tile's 16 async copies are the oldest outstanding; ASYNCcnt
    // decrements in order, so <=16 (next tile still in flight) is enough.
    if (have_next)
      asm volatile("s_wait_asynccnt 0x10" ::: "memory");
    else
      asm volatile("s_wait_asynccnt 0x0" ::: "memory");
    const __bf16* vsrc = &lds_v[ti & 1][0];
#pragma unroll
    for (int j = 0; j < 8; ++j) {              // PV: 8 d-subtiles of 16
      v16bf vfrag = *(const v16bf*)(vsrc + (j * 16 + l16) * 32 + half * 16);
      acc[j] = wmma_bf16(pt.v, vfrag, acc[j]);
    }
#else
#pragma unroll
    for (int j = 0; j < 8; ++j) {              // PV: 8 d-subtiles of 16
      v16bf vfrag =
          *(const v16bf*)(vbase + (size_t)(j * 16 + l16) * S_ + k0 + half * 16);
      acc[j] = wmma_bf16(pt.v, vfrag, acc[j]);
    }
#endif
    __syncthreads();
  }

  float inv[8];
#pragma unroll
  for (int g = 0; g < 8; ++g) inv[g] = 1.f / lr[g];
#pragma unroll
  for (int j = 0; j < 8; ++j)
#pragma unroll
    for (int g = 0; g < 8; ++g) {
      int srow = q0 + half * 8 + g;
      int d = j * 16 + l16;
      oout[(size_t)(b * S_ + srow) * HD_ + h * D_ + d] =
          __builtin_bit_cast(unsigned short, f2bf(acc[j][g] * inv[g]));
    }
}

// ---------------------------------------------------------------------------
extern "C" void kernel_launch(void* const* d_in, const int* in_sizes, int n_in,
                              void* d_out, int out_size, void* d_ws, size_t ws_size,
                              hipStream_t stream) {
  (void)in_sizes; (void)n_in; (void)out_size; (void)ws_size;

  const float* x    = (const float*)d_in[0];
  const float* mask = (const float*)d_in[1];
  const float* Wf[4] = {(const float*)d_in[2],  (const float*)d_in[6],
                        (const float*)d_in[10], (const float*)d_in[14]};
  const float* bf[4] = {(const float*)d_in[3],  (const float*)d_in[7],
                        (const float*)d_in[11], (const float*)d_in[15]};
  const float* Af[4] = {(const float*)d_in[4],  (const float*)d_in[8],
                        (const float*)d_in[12], (const float*)d_in[16]};
  const float* Bf[4] = {(const float*)d_in[5],  (const float*)d_in[9],
                        (const float*)d_in[13], (const float*)d_in[17]};

  char* w = (char*)d_ws;
  size_t off = 0;
  auto alloc = [&](size_t bytes) -> void* {
    void* p = w + off;
    off += (bytes + 255) & ~(size_t)255;
    return p;
  };

  unsigned short* xb = (unsigned short*)alloc((size_t)M_ * E_ * 2);
  unsigned short *Wb[4], *Ab[4], *Bb[4];
  for (int i = 0; i < 4; ++i) {
    Wb[i] = (unsigned short*)alloc((size_t)HD_ * E_ * 2);
    Ab[i] = (unsigned short*)alloc((size_t)NR_ * E_ * 2);
    Bb[i] = (unsigned short*)alloc((size_t)HD_ * NR_ * 2);
  }
  float* qf   = (float*)alloc((size_t)M_ * HD_ * 4);
  float* kf   = (float*)alloc((size_t)M_ * HD_ * 4);
  float* vf   = (float*)alloc((size_t)M_ * HD_ * 4);
  float* hbuf = (float*)alloc((size_t)M_ * NR_ * 4);
  unsigned short* hb  = (unsigned short*)alloc((size_t)M_ * NR_ * 2);
  unsigned short* qbb = (unsigned short*)alloc((size_t)M_ * HD_ * 2);
  unsigned short* kbb = (unsigned short*)alloc((size_t)M_ * HD_ * 2);
  unsigned short* vtb = (unsigned short*)alloc((size_t)M_ * HD_ * 2);
  unsigned short* obb = (unsigned short*)alloc((size_t)M_ * HD_ * 2);

  // precision conversion (weights then live in L2: ~37 MB bf16 << 192 MB)
  cvt_f32_bf16<<<(M_ * E_) / 256, 256, 0, stream>>>(x, xb);
  for (int i = 0; i < 4; ++i) {
    cvt_f32_bf16<<<(HD_ * E_) / 256, 256, 0, stream>>>(Wf[i], Wb[i]);
    cvt_f32_bf16<<<(NR_ * E_) / 256, 256, 0, stream>>>(Af[i], Ab[i]);
    cvt_f32_bf16<<<(HD_ * NR_) / 256, 256, 0, stream>>>(Bf[i], Bb[i]);
  }

  int gd = (M_ / 16) * (HD_ / 64) / 4;   // dense GEMM blocks
  int gl = (M_ / 16) * (NR_ / 64) / 4;   // LoRA-down GEMM blocks

  float* outs[3] = {qf, kf, vf};
  for (int p = 0; p < 3; ++p) {
    gemm_bf16_wmma<<<gd, 128, 0, stream>>>(xb, Wb[p], bf[p], outs[p], M_, HD_, E_, 0);
    gemm_bf16_wmma<<<gl, 128, 0, stream>>>(xb, Ab[p], nullptr, hbuf, M_, NR_, E_, 0);
    mask_scale_cvt<<<(M_ * NR_) / 256, 256, 0, stream>>>(hbuf, mask, hb);
    gemm_bf16_wmma<<<gd, 128, 0, stream>>>(hb, Bb[p], nullptr, outs[p], M_, HD_, NR_, 1);
  }

  rope_cvt<<<(M_ * HD_) / 256, 256, 0, stream>>>(qf, qbb);
  rope_cvt<<<(M_ * HD_) / 256, 256, 0, stream>>>(kf, kbb);
  v_transpose_cvt<<<(M_ * HD_) / 256, 256, 0, stream>>>(vf, vtb);

  flash_attn<<<B_ * H_ * (S_ / 16), 32, 0, stream>>>(qbb, kbb, vtb, obb);

  float* out = (float*)d_out;
  gemm_bf16_wmma<<<gd, 128, 0, stream>>>(obb, Wb[3], bf[3], out, M_, HD_, E_, 0);
  gemm_bf16_wmma<<<gl, 128, 0, stream>>>(obb, Ab[3], nullptr, hbuf, M_, NR_, E_, 0);
  mask_scale_cvt<<<(M_ * NR_) / 256, 256, 0, stream>>>(hbuf, mask, hb);
  gemm_bf16_wmma<<<gd, 128, 0, stream>>>(hb, Bb[3], nullptr, out, M_, HD_, NR_, 1);
}
